// Greed_Hinge_87694642250037
// MI455X (gfx1250) — matrix operand order repaired and checked by reading, not compiled, not run on hardware
//
#include <hip/hip_runtime.h>
#include <math.h>

#define N_NODES   50000
#define N_EDGES   640000
#define N_GRAPHS  1000
#define IN_DIM    32
#define HID       128
#define N_LAYERS  4
#define CAT_DIM   (HID * (N_LAYERS + 1))   // 640
#define OUT_DIM   128

typedef __attribute__((ext_vector_type(16))) __bf16 v16bf;
typedef __attribute__((ext_vector_type(8)))  float  v8f;

// ---------------- helpers ----------------

__device__ __forceinline__ unsigned short f2bf(float f) {
    unsigned int b = __float_as_uint(f);
    b += 0x7FFFu + ((b >> 16) & 1u);          // round-to-nearest-even
    return (unsigned short)(b >> 16);
}

union Frag16 { v16bf v; unsigned short u[16]; };

// ---------------- small elementwise kernels ----------------

__global__ void k_zero(float* p, int n) {
    int i = blockIdx.x * blockDim.x + threadIdx.x;
    if (i < n) p[i] = 0.0f;
}

__global__ void k_cvt(const float* __restrict__ a, unsigned short* __restrict__ o, int n) {
    int i = blockIdx.x * blockDim.x + threadIdx.x;
    if (i < n) o[i] = f2bf(a[i]);
}

__global__ void k_addcvt(const float* __restrict__ a, const float* __restrict__ b,
                         unsigned short* __restrict__ o, int n) {
    int i = blockIdx.x * blockDim.x + threadIdx.x;
    if (i < n) o[i] = f2bf(a[i] + b[i]);
}

// Pack a row-major fp32 weight matrix [K=Ksteps*32, 128] into the CDNA5
// 16-bit B-matrix lane layout, tiled as [tileN(8)][kstep][lane(32)][half(16)].
// Per-lane K index: lanes 0-15 hold K {0..7, 16..23}, lanes 16-31 hold K {8..15, 24..31}
// (per ISA 05_wmma B layout); column N = tileN*16 + (lane & 15).
__global__ void k_pack_b(const float* __restrict__ W, unsigned short* __restrict__ dst, int Ksteps) {
    int i = blockIdx.x * blockDim.x + threadIdx.x;
    int total = 8 * Ksteps * 512;
    if (i >= total) return;
    int hi   = i & 15;
    int lane = (i >> 4) & 31;
    int ks   = (i >> 9) % Ksteps;
    int tn   = (i >> 9) / Ksteps;
    int k = ks * 32 + ((lane & 16) ? 8 : 0) + (hi & 7) + ((hi & 8) ? 16 : 0);
    int n = tn * 16 + (lane & 15);
    dst[i] = f2bf(W[(size_t)k * HID + n]);
}

// agg[dst[e]] += x[src[e]]  (128 floats per edge, 4 floats per thread)
__global__ void k_scatter(const float* __restrict__ x, const int* __restrict__ src,
                          const int* __restrict__ dst, float* __restrict__ agg, int nE) {
    int t = blockIdx.x * blockDim.x + threadIdx.x;
    if (t >= nE * 32) return;
    int e = t >> 5;
    int c = (t & 31) << 2;
    int s = src[e], d = dst[e];
    const float4 v = *reinterpret_cast<const float4*>(x + (size_t)s * HID + c);
    float* o = agg + (size_t)d * HID + c;
    atomicAdd(o + 0, v.x);
    atomicAdd(o + 1, v.y);
    atomicAdd(o + 2, v.z);
    atomicAdd(o + 3, v.w);
}

// pooled[batch[node]][slot*128 + d] += act[node][d]
__global__ void k_pool(const float* __restrict__ act, const int* __restrict__ batch,
                       float* __restrict__ pooled, int slot) {
    int t = blockIdx.x * blockDim.x + threadIdx.x;
    if (t >= N_NODES * HID) return;
    int node = t >> 7;
    int d    = t & 127;
    atomicAdd(&pooled[(size_t)batch[node] * CAT_DIM + slot * HID + d], act[t]);
}

// ---------------- WMMA GEMM: C[M,128] = A[M,K] @ Wpacked + bias, fused epilogue ----------------
// One wave per 16x16 C tile. K = Ksteps*32. A is bf16 row-major with leading dim lda.
__global__ void k_gemm(const unsigned short* __restrict__ A, int lda, int Ksteps,
                       const unsigned short* __restrict__ Bp, const float* __restrict__ bias,
                       const float* __restrict__ residIn, float* __restrict__ residOut,
                       float* __restrict__ Cf, unsigned short* __restrict__ Cbf,
                       int M, int doRelu) {
    int wave = (blockIdx.x * blockDim.x + threadIdx.x) >> 5;
    int lane = threadIdx.x & 31;
    int Mtiles = (M + 15) >> 4;
    if (wave >= Mtiles * 8) return;
    int tm = wave >> 3;
    int tn = wave & 7;

    int rowA = tm * 16 + (lane & 15);
    if (rowA >= M) rowA = M - 1;                    // clamp reads; stores guarded below
    // A lane layout: lanes 0-15 -> K {0..7, 16..23}; lanes 16-31 -> K {8..15, 24..31}
    const unsigned short* Arow = A + (size_t)rowA * lda + ((lane & 16) ? 8 : 0);
    const unsigned short* Bl   = Bp + ((size_t)(tn * Ksteps) * 32 + lane) * 16;

    v8f c = {};
    for (int ks = 0; ks < Ksteps; ++ks) {
        Frag16 a, b;
        const uint4* ap = reinterpret_cast<const uint4*>(Arow + ks * 32);
        *reinterpret_cast<uint4*>(&a.u[0]) = ap[0];     // K base + 0..7
        *reinterpret_cast<uint4*>(&a.u[8]) = ap[2];     // K base + 16..23
        const uint4* bp = reinterpret_cast<const uint4*>(Bl + (size_t)ks * 512);
        *reinterpret_cast<uint4*>(&b.u[0]) = bp[0];
        *reinterpret_cast<uint4*>(&b.u[8]) = bp[1];
        c = __builtin_amdgcn_wmma_f32_16x16x32_bf16(false, a.v, false, b.v,
                                                    (short)0, c, false, false);
    }

    int col = tn * 16 + (lane & 15);
    float bv = bias ? bias[col] : 0.0f;
    int rbase = tm * 16 + ((lane & 16) ? 8 : 0);     // C/D layout: VGPR r -> row r (+8 upper half)
#pragma unroll
    for (int r = 0; r < 8; ++r) {
        int row = rbase + r;
        if (row < M) {
            size_t idx = (size_t)row * HID + col;
            float v = c[r] + bv;
            if (residIn)  v += residIn[idx];
            if (residOut) residOut[idx] = v;        // pre-relu value (residual carry)
            if (doRelu)   v = fmaxf(v, 0.0f);
            if (Cf)  Cf[idx]  = v;
            if (Cbf) Cbf[idx] = f2bf(v);
        }
    }
}

// ---------------- final asymmetric relu-L2 norm (one wave per graph) ----------------
__global__ void k_norm(const float* __restrict__ gx, const float* __restrict__ hx,
                       float* __restrict__ out) {
    int wv   = (blockIdx.x * blockDim.x + threadIdx.x) >> 5;
    int lane = threadIdx.x & 31;
    if (wv >= N_GRAPHS) return;
    float snd = 0.f, sni = 0.f, sed = 0.f, sei = 0.f;
#pragma unroll
    for (int j = 0; j < 4; ++j) {
        int d = lane * 4 + j;
        float diff = gx[(size_t)wv * 128 + d] - hx[(size_t)wv * 128 + d];
        float del = diff > 0.f ?  diff : 0.f; del *= del;
        float ins = diff < 0.f ? -diff : 0.f; ins *= ins;
        if (d < 64) { snd += del; sni += ins; }
        else        { sed += del; sei += ins; }
    }
#pragma unroll
    for (int m = 16; m >= 1; m >>= 1) {
        snd += __shfl_xor(snd, m, 32);
        sni += __shfl_xor(sni, m, 32);
        sed += __shfl_xor(sed, m, 32);
        sei += __shfl_xor(sei, m, 32);
    }
    if (lane == 0)
        out[wv] = sqrtf(snd) + sqrtf(sni) + sqrtf(sed) + sqrtf(sei);
}

// ---------------- host orchestration ----------------

extern "C" void kernel_launch(void* const* d_in, const int* in_sizes, int n_in,
                              void* d_out, int out_size, void* d_ws, size_t ws_size,
                              hipStream_t stream) {
    (void)in_sizes; (void)n_in; (void)out_size; (void)ws_size;

    const float* g_x     = (const float*)d_in[0];
    const int*   g_eix   = (const int*)  d_in[1];
    const int*   g_batch = (const int*)  d_in[2];
    const float* h_x     = (const float*)d_in[3];
    const int*   h_eix   = (const int*)  d_in[4];
    const int*   h_batch = (const int*)  d_in[5];
    const float* pre_w   = (const float*)d_in[6];
    const float* pre_b   = (const float*)d_in[7];
    const float* gin_w1  = (const float*)d_in[8];
    const float* gin_b1  = (const float*)d_in[9];
    const float* gin_w2  = (const float*)d_in[10];
    const float* gin_b2  = (const float*)d_in[11];
    const float* post_w1 = (const float*)d_in[12];
    const float* post_b1 = (const float*)d_in[13];
    const float* post_w2 = (const float*)d_in[14];
    const float* post_b2 = (const float*)d_in[15];
    float* out = (float*)d_out;

    // bump allocator over d_ws
    char* ws = (char*)d_ws;
    size_t off = 0;
    auto alloc = [&](size_t bytes) -> char* {
        char* p = ws + off;
        off += (bytes + 255) & ~(size_t)255;
        return p;
    };

    unsigned short* preWp = (unsigned short*)alloc((size_t)8 * 1  * 512 * 2);
    unsigned short* w1p   = (unsigned short*)alloc((size_t)N_LAYERS * 8 * 4 * 512 * 2);
    unsigned short* w2p   = (unsigned short*)alloc((size_t)N_LAYERS * 8 * 4 * 512 * 2);
    unsigned short* pw1p  = (unsigned short*)alloc((size_t)8 * 20 * 512 * 2);
    unsigned short* pw2p  = (unsigned short*)alloc((size_t)8 * 4  * 512 * 2);
    float*          x_f   = (float*)alloc((size_t)N_NODES * HID * 4);
    float*          xres  = (float*)alloc((size_t)N_NODES * HID * 4);
    float*          agg   = (float*)alloc((size_t)N_NODES * HID * 4);
    unsigned short* abf   = (unsigned short*)alloc((size_t)N_NODES * HID * 2);
    unsigned short* bbf   = (unsigned short*)alloc((size_t)N_NODES * HID * 2);
    float*          pooled   = (float*)alloc((size_t)N_GRAPHS * CAT_DIM * 4);
    unsigned short* pooledbf = (unsigned short*)alloc((size_t)N_GRAPHS * CAT_DIM * 2);
    unsigned short* phbf     = (unsigned short*)alloc((size_t)N_GRAPHS * HID * 2);
    float*          gxbuf    = (float*)alloc((size_t)N_GRAPHS * HID * 4);
    float*          hxbuf    = (float*)alloc((size_t)N_GRAPHS * HID * 4);

    // --- pack all weight matrices into WMMA B layout (bf16), once per call ---
    auto pack = [&](const float* W, unsigned short* dst, int Ksteps) {
        int total = 8 * Ksteps * 512;
        k_pack_b<<<(total + 255) / 256, 256, 0, stream>>>(W, dst, Ksteps);
    };
    pack(pre_w, preWp, 1);
    for (int i = 0; i < N_LAYERS; ++i) {
        pack(gin_w1 + (size_t)i * HID * HID, w1p + (size_t)i * 16384, 4);
        pack(gin_w2 + (size_t)i * HID * HID, w2p + (size_t)i * 16384, 4);
    }
    pack(post_w1, pw1p, 20);
    pack(post_w2, pw2p, 4);

    const int bigGemmBlocks  = ((N_NODES / 16) * 8) / 4;          // 3125*8 waves / 4 per block
    const int poolGemmBlocks = (((N_GRAPHS + 15) / 16) * 8 + 3) / 4;
    const int nodeElems = N_NODES * HID;

    auto runGraph = [&](const float* xin, const int* eix, const int* batch, float* outEmb) {
        // pre layer: x = xin @ pre_w + pre_b ; xres = x
        k_cvt<<<(N_NODES * IN_DIM + 255) / 256, 256, 0, stream>>>(xin, abf, N_NODES * IN_DIM);
        k_gemm<<<bigGemmBlocks, 128, 0, stream>>>(abf, IN_DIM, 1, preWp, pre_b,
                                                  nullptr, xres, x_f, nullptr, N_NODES, 0);
        k_zero<<<(N_GRAPHS * CAT_DIM + 255) / 256, 256, 0, stream>>>(pooled, N_GRAPHS * CAT_DIM);
        k_pool<<<(nodeElems + 255) / 256, 256, 0, stream>>>(x_f, batch, pooled, 0);

        for (int i = 0; i < N_LAYERS; ++i) {
            k_zero<<<(nodeElems + 255) / 256, 256, 0, stream>>>(agg, nodeElems);
            k_scatter<<<(N_EDGES * 32 + 255) / 256, 256, 0, stream>>>(x_f, eix, eix + N_EDGES,
                                                                      agg, N_EDGES);
            k_addcvt<<<(nodeElems + 255) / 256, 256, 0, stream>>>(agg, x_f, abf, nodeElems);
            // t = relu((agg+x) @ w1 + b1)  -> bf16 only
            k_gemm<<<bigGemmBlocks, 128, 0, stream>>>(abf, HID, 4,
                                                      w1p + (size_t)i * 16384, gin_b1 + i * HID,
                                                      nullptr, nullptr, nullptr, bbf, N_NODES, 1);
            // h = t @ w2 + b2 (+xres on odd layers, xres <- pre-relu h) ; x = relu(h)
            bool odd = (i & 1) != 0;
            k_gemm<<<bigGemmBlocks, 128, 0, stream>>>(bbf, HID, 4,
                                                      w2p + (size_t)i * 16384, gin_b2 + i * HID,
                                                      odd ? xres : nullptr, odd ? xres : nullptr,
                                                      x_f, nullptr, N_NODES, 1);
            k_pool<<<(nodeElems + 255) / 256, 256, 0, stream>>>(x_f, batch, pooled, i + 1);
        }

        // post MLP on pooled [1000, 640]
        k_cvt<<<(N_GRAPHS * CAT_DIM + 255) / 256, 256, 0, stream>>>(pooled, pooledbf,
                                                                    N_GRAPHS * CAT_DIM);
        k_gemm<<<poolGemmBlocks, 128, 0, stream>>>(pooledbf, CAT_DIM, 20, pw1p, post_b1,
                                                   nullptr, nullptr, nullptr, phbf, N_GRAPHS, 1);
        k_gemm<<<poolGemmBlocks, 128, 0, stream>>>(phbf, HID, 4, pw2p, post_b2,
                                                   nullptr, nullptr, outEmb, nullptr, N_GRAPHS, 0);
    };

    runGraph(g_x, g_eix, g_batch, gxbuf);
    runGraph(h_x, h_eix, h_batch, hxbuf);

    // 1000 graphs, one wave each, 4 waves per 128-thread block
    k_norm<<<(N_GRAPHS + 3) / 4, 128, 0, stream>>>(gxbuf, hxbuf, out);
}
